// MultiHeadHierarchicalAttention_54906861912656
// MI455X (gfx1250) — matrix operand order repaired and checked by reading, not compiled, not run on hardware
//
#include <hip/hip_runtime.h>

// ---------------- problem constants ----------------
#define B_   4
#define LQ_  512
#define NB_  32
#define NT_  256
#define DM_  512
#define H_   8
#define DK_  64
#define DV_  64
#define SCALE_ 0.125f   // 1/sqrt(64)

typedef __attribute__((ext_vector_type(16))) __bf16 v16bf;
typedef __attribute__((ext_vector_type(8)))  float  v8f;

// ---------------- WMMA helpers (CDNA5, wave32) ----------------
__device__ __forceinline__ v8f wmma_bf16(v16bf a, v16bf b, v8f c) {
  // (neg_a, A, neg_b, B, c_mod, C, reuse_a, reuse_b)
  return __builtin_amdgcn_wmma_f32_16x16x32_bf16(false, a, false, b, (short)0, c, false, false);
}

// A fragment (16x32 bf16, M x K): lane holds row m=lane&15.
// Per ISA 7.12.2: lanes 0-15 hold k {0..7,16..23}; lanes 16-31 hold k {8..15,24..31}.
__device__ __forceinline__ v16bf frag_a_bf16(const __bf16* p, int ld) {
  const int lane = threadIdx.x & 31;
  const __bf16* row = p + (size_t)(lane & 15) * ld + ((lane >> 4) << 3);
  v16bf f;
#pragma unroll
  for (int i = 0; i < 8; ++i) { f[i] = row[i]; f[i + 8] = row[16 + i]; }
  return f;
}

__device__ __forceinline__ v16bf frag_a_f32(const float* p, int ld) {
  const int lane = threadIdx.x & 31;
  const float* row = p + (size_t)(lane & 15) * ld + ((lane >> 4) << 3);
  v16bf f;
#pragma unroll
  for (int i = 0; i < 8; ++i) { f[i] = (__bf16)row[i]; f[i + 8] = (__bf16)row[16 + i]; }
  return f;
}

// B fragment (32x16 bf16, K x N) loaded from an N-major buffer Bt[n][k]:
// lane holds column n=lane&15; lanes 0-15 hold k {0..7,16..23}; lanes 16-31 hold k {8..15,24..31}.
__device__ __forceinline__ v16bf frag_b_bf16(const __bf16* p, int ld) {
  const int lane = threadIdx.x & 31;
  const __bf16* row = p + (size_t)(lane & 15) * ld + ((lane >> 4) << 3);
  v16bf f;
#pragma unroll
  for (int i = 0; i < 8; ++i) { f[i] = row[i]; f[i + 8] = row[16 + i]; }
  return f;
}

// ---------------- CDNA5 async global->LDS copy (ASYNCcnt path) ----------------
// GV mode: per-lane 64-bit global address in VGPR pair, per-lane LDS byte
// address in a VGPR; 16 bytes moved per lane, tracked with ASYNCcnt.
__device__ __forceinline__ void async_ld_b128(unsigned lds_byte_off, const void* gaddr) {
  asm volatile("global_load_async_to_lds_b128 %0, %1, off"
               :: "v"(lds_byte_off), "v"(gaddr)
               : "memory");
}
__device__ __forceinline__ void wait_async0() {
  asm volatile("s_wait_asynccnt 0x0" ::: "memory");
}

// ---------------- weight transpose + bf16 convert ----------------
// Wt[n*K + k] = (bf16) W[k*N + n]
__global__ __launch_bounds__(256) void transpose_bf16_kernel(
    const float* __restrict__ W, __bf16* __restrict__ Wt, int K, int N) {
  size_t idx = (size_t)blockIdx.x * 256 + threadIdx.x;
  if (idx >= (size_t)K * N) return;
  int k = (int)(idx % K);
  int n = (int)(idx / K);
  Wt[idx] = (__bf16)W[(size_t)k * N + n];
}

// ---------------- generic WMMA GEMM: C[M][N] = A[M][K] * Bt[N][K]^T (+bias)(+accum) ----------------
// grid = (N/128, M/64); block = 256 (8 waves, 2x4); each wave owns a 32x32 tile (4 wmma accs)
template <bool OUT_BF16, bool ACCUM, bool BIAS>
__global__ __launch_bounds__(256) void gemm_kernel(
    const float* __restrict__ A, int lda,
    const __bf16* __restrict__ Bt, int ldb,
    const float* __restrict__ bias,
    void* __restrict__ Cv, int ldc, int K) {
  const int lane = threadIdx.x & 31;
  const int wave = threadIdx.x >> 5;
  const int m0 = blockIdx.y * 64 + (wave & 1) * 32;
  const int n0 = blockIdx.x * 128 + (wave >> 1) * 32;

  v8f acc00 = {}, acc01 = {}, acc10 = {}, acc11 = {};
  for (int k = 0; k < K; k += 32) {
    v16bf a0 = frag_a_f32(A + (size_t)m0 * lda + k, lda);
    v16bf a1 = frag_a_f32(A + (size_t)(m0 + 16) * lda + k, lda);
    v16bf b0 = frag_b_bf16(Bt + (size_t)n0 * ldb + k, ldb);
    v16bf b1 = frag_b_bf16(Bt + (size_t)(n0 + 16) * ldb + k, ldb);
    acc00 = wmma_bf16(a0, b0, acc00);
    acc01 = wmma_bf16(a0, b1, acc01);
    acc10 = wmma_bf16(a1, b0, acc10);
    acc11 = wmma_bf16(a1, b1, acc11);
  }

  v8f accs[2][2] = {{acc00, acc01}, {acc10, acc11}};
#pragma unroll
  for (int ti = 0; ti < 2; ++ti) {
#pragma unroll
    for (int tj = 0; tj < 2; ++tj) {
      v8f acc = accs[ti][tj];
      const int n = n0 + tj * 16 + (lane & 15);
      if (BIAS) {
        float bv = bias[n];
#pragma unroll
        for (int r = 0; r < 8; ++r) acc[r] += bv;
      }
      const int mbase = m0 + ti * 16 + ((lane >> 4) << 3);
      if (OUT_BF16) {
        __bf16* C = (__bf16*)Cv;
#pragma unroll
        for (int r = 0; r < 8; ++r) C[(size_t)(mbase + r) * ldc + n] = (__bf16)acc[r];
      } else {
        float* C = (float*)Cv;
#pragma unroll
        for (int r = 0; r < 8; ++r) {
          size_t i = (size_t)(mbase + r) * ldc + n;
          float v = acc[r];
          if (ACCUM) v += C[i];
          C[i] = v;
        }
      }
    }
  }
}

// ---------------- sentence-level attention (small; VALU) ----------------
__global__ __launch_bounds__(256) void sent_logits_kernel(
    const __bf16* __restrict__ qs, const __bf16* __restrict__ ks,
    float* __restrict__ attn) {
  int idx = blockIdx.x * 256 + threadIdx.x;  // [b][h][q][n], n fastest
  if (idx >= B_ * H_ * LQ_ * NB_) return;
  int n = idx & 31;
  int q = (idx >> 5) & 511;
  int h = (idx >> 14) & 7;
  int b = idx >> 17;
  const __bf16* qp = qs + ((size_t)(b * LQ_ + q)) * DM_ + h * DK_;
  const __bf16* kp = ks + ((size_t)(b * NB_ + n)) * DM_ + h * DK_;
  float s = 0.f;
#pragma unroll 8
  for (int d = 0; d < DK_; ++d) s += (float)qp[d] * (float)kp[d];
  attn[idx] = s * SCALE_;
}

__global__ __launch_bounds__(256) void sent_softmax_kernel(float* __restrict__ attn) {
  int row = blockIdx.x * 256 + threadIdx.x;  // B*H*LQ rows of 32
  if (row >= B_ * H_ * LQ_) return;
  float* p = attn + (size_t)row * NB_;
  float mx = -1e30f;
#pragma unroll
  for (int j = 0; j < NB_; ++j) mx = fmaxf(mx, p[j]);
  float sum = 0.f;
#pragma unroll
  for (int j = 0; j < NB_; ++j) { float e = __expf(p[j] - mx); p[j] = e; sum += e; }
  float inv = 1.f / sum;
#pragma unroll
  for (int j = 0; j < NB_; ++j) p[j] *= inv;
}

__global__ __launch_bounds__(256) void sent_ctx_kernel(
    const float* __restrict__ attn, const __bf16* __restrict__ vs,
    float* __restrict__ ctx_s) {
  int idx = blockIdx.x * 256 + threadIdx.x;  // [b][q][h][d], d fastest
  if (idx >= B_ * LQ_ * H_ * DV_) return;
  int d = idx & 63;
  int h = (idx >> 6) & 7;
  int q = (idx >> 9) & 511;
  int b = idx >> 18;
  const float* ap = attn + (((size_t)(b * H_ + h) * LQ_) + q) * NB_;
  const __bf16* vp = vs + ((size_t)(b * NB_)) * DM_ + h * DV_ + d;
  float s = 0.f;
#pragma unroll
  for (int n = 0; n < NB_; ++n) s += ap[n] * (float)vp[(size_t)n * DM_];
  ctx_s[idx] = s;
}

// ---------------- token-level attention (WMMA, async-LDS staging) ----------------
// grid = (LQ/64, H, B); block = 256; dynamic LDS layout (byte offsets):
//   sQ  [64][64]  bf16 @ 0        (8 KB)
//   sK  [256][64] bf16 @ 8192     (32 KB)
//   sVt [64][256] bf16 @ 40960    (32 KB)
//   sS  [64][256] f32  @ 73728    (64 KB)
//   sP  [64][256] bf16 @ 139264   (32 KB)
#define OFF_Q  0u
#define OFF_K  8192u
#define OFF_VT 40960u
#define OFF_S  73728u
#define OFF_P  139264u
#define TOKEN_SMEM 172032

__global__ __launch_bounds__(256) void token_attn_kernel(
    const __bf16* __restrict__ qw, const __bf16* __restrict__ kw,
    const __bf16* __restrict__ vw, const float* __restrict__ attn_s,
    float* __restrict__ ctx_w) {
  extern __shared__ char smem[];
  __bf16* sQ  = (__bf16*)(smem + OFF_Q);
  __bf16* sK  = (__bf16*)(smem + OFF_K);
  __bf16* sVt = (__bf16*)(smem + OFF_VT);
  float*  sS  = (float*)(smem + OFF_S);
  __bf16* sP  = (__bf16*)(smem + OFF_P);

  const int qt = blockIdx.x, h = blockIdx.y, b = blockIdx.z;
  const int tid = threadIdx.x, lane = tid & 31, wave = tid >> 5;

  // stage Q tile (64 q-rows x 64 dk) via async global->LDS, once.
  // chunk i = r*8 + dblk covers row r, d = dblk*8..dblk*8+7 (16 bytes).
  for (int i = tid; i < 512; i += 256) {
    int r = i >> 3, dblk = i & 7;
    const __bf16* g = qw + ((size_t)(b * LQ_ + qt * 64 + r)) * DM_ + h * DK_ + dblk * 8;
    async_ld_b128(OFF_Q + (unsigned)i * 16u, g);
  }

  // each wave owns two adjacent ctx tiles: (mt, nt0) and (mt, nt0+1)
  const int t0 = wave * 2;
  const int mt = t0 >> 2;
  const int nt0 = t0 & 3;
  v8f acc0 = {};
  v8f acc1 = {};

  for (int n = 0; n < NB_; ++n) {
    __syncthreads();  // previous iteration done with sK/sVt/sP
    const size_t base = ((size_t)(b * NB_ + n) * NT_) * DM_ + h * DK_;

    // K tile: contiguous copy -> async global->LDS b128.
    // chunk j = t*8 + dblk covers token t, d = dblk*8..+7.
    for (int j = tid; j < 2048; j += 256) {
      int t = j >> 3, dblk = j & 7;
      const __bf16* g = kw + base + (size_t)t * DM_ + dblk * 8;
      async_ld_b128(OFF_K + (unsigned)j * 16u, g);
    }
    // V tile: needs transpose -> vectorized b128 read, b16 transposed LDS stores.
    // chunk j: t = j & 255, dblk = j >> 8 (keeps LDS stores of adjacent lanes adjacent)
    for (int j = tid; j < 2048; j += 256) {
      int t = j & 255, dblk = j >> 8;
      const __bf16* g = vw + base + (size_t)t * DM_ + dblk * 8;
      __bf16 tmp[8];
      *(ulonglong2*)tmp = *(const ulonglong2*)g;
#pragma unroll
      for (int s = 0; s < 8; ++s) sVt[(dblk * 8 + s) * NT_ + t] = tmp[s];
    }
    wait_async0();
    __syncthreads();

    // S = SCALE * Q @ K^T : 4 m-tiles x 16 t-tiles = 64 jobs, 8 per wave
    for (int job = wave; job < 64; job += 8) {
      int smt = job & 3, stt = job >> 2;
      v8f s = {};
#pragma unroll
      for (int k = 0; k < DK_; k += 32) {
        v16bf a = frag_a_bf16(sQ + smt * 16 * 64 + k, 64);
        v16bf bb = frag_b_bf16(sK + stt * 16 * 64 + k, 64);
        s = wmma_bf16(a, bb, s);
      }
      float* col = sS + (size_t)(smt * 16 + ((lane >> 4) << 3)) * NT_ + stt * 16 + (lane & 15);
#pragma unroll
      for (int r = 0; r < 8; ++r) col[(size_t)r * NT_] = s[r] * SCALE_;
    }
    __syncthreads();

    // per-row softmax over 256 tokens, weighted by sentence attention
    if (tid < 64) {
      float* row = sS + (size_t)tid * NT_;
      float mx = -1e30f;
      for (int j = 0; j < NT_; ++j) mx = fmaxf(mx, row[j]);
      float sum = 0.f;
      for (int j = 0; j < NT_; ++j) { float e = __expf(row[j] - mx); row[j] = e; sum += e; }
      float ws = attn_s[(((size_t)(b * H_ + h) * LQ_) + qt * 64 + tid) * NB_ + n];
      float sc = ws / sum;
      __bf16* prow = sP + (size_t)tid * NT_;
      for (int j = 0; j < NT_; ++j) prow[j] = (__bf16)(row[j] * sc);
    }
    __syncthreads();

    // ctx += P @ V  (K = 256 tokens)
#pragma unroll
    for (int k = 0; k < NT_; k += 32) {
      v16bf a  = frag_a_bf16(sP + mt * 16 * NT_ + k, NT_);
      v16bf b0 = frag_b_bf16(sVt + nt0 * 16 * NT_ + k, NT_);
      v16bf b1 = frag_b_bf16(sVt + (nt0 + 1) * 16 * NT_ + k, NT_);
      acc0 = wmma_bf16(a, b0, acc0);
      acc1 = wmma_bf16(a, b1, acc1);
    }
  }

  // store ctx_w[b, q, h*64 + d]
  const int mrow = qt * 64 + mt * 16 + ((lane >> 4) << 3);
  const int ncol = (lane & 15);
#pragma unroll
  for (int r = 0; r < 8; ++r) {
    size_t rowoff = (size_t)(b * LQ_ + mrow + r) * DM_ + h * DV_;
    ctx_w[rowoff + nt0 * 16 + ncol] = acc0[r];
    ctx_w[rowoff + (nt0 + 1) * 16 + ncol] = acc1[r];
  }
}

// ---------------- host launcher ----------------
extern "C" void kernel_launch(void* const* d_in, const int* in_sizes, int n_in,
                              void* d_out, int out_size, void* d_ws, size_t ws_size,
                              hipStream_t stream) {
  (void)in_sizes; (void)n_in; (void)out_size; (void)ws_size;

  const float* q    = (const float*)d_in[0];
  const float* k_w  = (const float*)d_in[1];
  const float* v_w  = (const float*)d_in[2];
  const float* k_s  = (const float*)d_in[3];
  const float* v_s  = (const float*)d_in[4];
  const float* W_qs = (const float*)d_in[5];   const float* b_qs = (const float*)d_in[6];
  const float* W_ks = (const float*)d_in[7];   const float* b_ks = (const float*)d_in[8];
  const float* W_vs = (const float*)d_in[9];   const float* b_vs = (const float*)d_in[10];
  const float* W_qw = (const float*)d_in[11];  const float* b_qw = (const float*)d_in[12];
  const float* W_kw = (const float*)d_in[13];  const float* b_kw = (const float*)d_in[14];
  const float* W_vw = (const float*)d_in[15];  const float* b_vw = (const float*)d_in[16];
  const float* W_fc1 = (const float*)d_in[17]; const float* b_fc1 = (const float*)d_in[18];
  const float* W_fc  = (const float*)d_in[19]; const float* b_fc  = (const float*)d_in[20];

  char* ws = (char*)d_ws;
  size_t off = 0;
  auto carve = [&](size_t bytes) { size_t o = off; off += (bytes + 255) & ~(size_t)255; return o; };

  const size_t WB = (size_t)DM_ * DM_ * 2;  // 512x512 bf16
  __bf16* wt_qs  = (__bf16*)(ws + carve(WB));
  __bf16* wt_ks  = (__bf16*)(ws + carve(WB));
  __bf16* wt_vs  = (__bf16*)(ws + carve(WB));
  __bf16* wt_qw  = (__bf16*)(ws + carve(WB));
  __bf16* wt_kw  = (__bf16*)(ws + carve(WB));
  __bf16* wt_vw  = (__bf16*)(ws + carve(WB));
  __bf16* wt_fc1 = (__bf16*)(ws + carve(WB));
  __bf16* wt_fc  = (__bf16*)(ws + carve((size_t)DM_ * 2 * DM_ * 2));  // [512][1024] bf16

  const size_t MQ = (size_t)B_ * LQ_;       // 2048
  const size_t MS = (size_t)B_ * NB_;       // 128
  const size_t MW = (size_t)B_ * NB_ * NT_; // 32768

  __bf16* qs_b = (__bf16*)(ws + carve(MQ * DM_ * 2));
  __bf16* ks_b = (__bf16*)(ws + carve(MS * DM_ * 2));
  __bf16* vs_b = (__bf16*)(ws + carve(MS * DM_ * 2));
  __bf16* qw_b = (__bf16*)(ws + carve(MQ * DM_ * 2));
  __bf16* kw_b = (__bf16*)(ws + carve(MW * DM_ * 2));
  __bf16* vw_b = (__bf16*)(ws + carve(MW * DM_ * 2));
  float* attn  = (float*)(ws + carve((size_t)B_ * H_ * LQ_ * NB_ * 4));
  float* ctx_s = (float*)(ws + carve(MQ * DM_ * 4));
  float* tmp1  = (float*)(ws + carve(MQ * DM_ * 4));
  float* ctx_w = (float*)(ws + carve(MQ * DM_ * 4));

  float* out = (float*)d_out;

  // 1) transpose + convert weights
  const int TGRID = (int)((WB / 2 + 255) / 256);
  transpose_bf16_kernel<<<TGRID, 256, 0, stream>>>(W_qs, wt_qs, DM_, DM_);
  transpose_bf16_kernel<<<TGRID, 256, 0, stream>>>(W_ks, wt_ks, DM_, DM_);
  transpose_bf16_kernel<<<TGRID, 256, 0, stream>>>(W_vs, wt_vs, DM_, DM_);
  transpose_bf16_kernel<<<TGRID, 256, 0, stream>>>(W_qw, wt_qw, DM_, DM_);
  transpose_bf16_kernel<<<TGRID, 256, 0, stream>>>(W_kw, wt_kw, DM_, DM_);
  transpose_bf16_kernel<<<TGRID, 256, 0, stream>>>(W_vw, wt_vw, DM_, DM_);
  transpose_bf16_kernel<<<TGRID, 256, 0, stream>>>(W_fc1, wt_fc1, DM_, DM_);
  transpose_bf16_kernel<<<2 * TGRID, 256, 0, stream>>>(W_fc, wt_fc, 2 * DM_, DM_);

  // 2) projections (f32 A, bf16 out); grid = (N/128, M/64)
  dim3 blk(256);
  gemm_kernel<true, false, true><<<dim3(DM_ / 128, (unsigned)(MQ / 64)), blk, 0, stream>>>(
      q, DM_, wt_qs, DM_, b_qs, qs_b, DM_, DM_);
  gemm_kernel<true, false, true><<<dim3(DM_ / 128, (unsigned)(MS / 64)), blk, 0, stream>>>(
      k_s, DM_, wt_ks, DM_, b_ks, ks_b, DM_, DM_);
  gemm_kernel<true, false, true><<<dim3(DM_ / 128, (unsigned)(MS / 64)), blk, 0, stream>>>(
      v_s, DM_, wt_vs, DM_, b_vs, vs_b, DM_, DM_);
  gemm_kernel<true, false, true><<<dim3(DM_ / 128, (unsigned)(MQ / 64)), blk, 0, stream>>>(
      q, DM_, wt_qw, DM_, b_qw, qw_b, DM_, DM_);
  gemm_kernel<true, false, true><<<dim3(DM_ / 128, (unsigned)(MW / 64)), blk, 0, stream>>>(
      k_w, DM_, wt_kw, DM_, b_kw, kw_b, DM_, DM_);
  gemm_kernel<true, false, true><<<dim3(DM_ / 128, (unsigned)(MW / 64)), blk, 0, stream>>>(
      v_w, DM_, wt_vw, DM_, b_vw, vw_b, DM_, DM_);

  // 3) sentence attention
  sent_logits_kernel<<<(B_ * H_ * LQ_ * NB_ + 255) / 256, blk, 0, stream>>>(qs_b, ks_b, attn);
  sent_softmax_kernel<<<(B_ * H_ * LQ_ + 255) / 256, blk, 0, stream>>>(attn);
  sent_ctx_kernel<<<(B_ * LQ_ * H_ * DV_ + 255) / 256, blk, 0, stream>>>(attn, vs_b, ctx_s);

  // 4) ctx_s @ W_fc1 + b_fc1 -> tmp1 (f32)
  gemm_kernel<false, false, true><<<dim3(DM_ / 128, (unsigned)(MQ / 64)), blk, 0, stream>>>(
      ctx_s, DM_, wt_fc1, DM_, b_fc1, tmp1, DM_, DM_);

  // 5) token attention -> ctx_w (f32)
  token_attn_kernel<<<dim3(LQ_ / 64, H_, B_), blk, TOKEN_SMEM, stream>>>(
      qw_b, kw_b, vw_b, attn, ctx_w);

  // 6) out = tmp1 @ W_fc[0:512] + b_fc ; out += ctx_w @ W_fc[512:1024]
  gemm_kernel<false, false, true><<<dim3(DM_ / 128, (unsigned)(MQ / 64)), blk, 0, stream>>>(
      tmp1, DM_, wt_fc, 2 * DM_, b_fc, out, DM_, DM_);
  gemm_kernel<false, true, false><<<dim3(DM_ / 128, (unsigned)(MQ / 64)), blk, 0, stream>>>(
      ctx_w, DM_, wt_fc + DM_, 2 * DM_, nullptr, out, DM_, DM_);
}
